// TriPlaneINR_50457275793905
// MI455X (gfx1250) — matrix operand order repaired and checked by reading, not compiled, not run on hardware
//
#include <hip/hip_runtime.h>
#include <hip/hip_bf16.h>
#include <math.h>

typedef __attribute__((ext_vector_type(16))) _Float16 v16h;
typedef __attribute__((ext_vector_type(8)))  _Float16 v8h;
typedef __attribute__((ext_vector_type(8)))  float    v8f;

#define HID 64
#define ROWS 72          // LDS row stride in f16 elements (144B, 16B-aligned, bank-spread)
#define RES_ 256
#define PLANE_CH_STRIDE 65536            // 256*256 (channel-major source planes)
#define PIXELS 65536                     // 256*256
#define TPLANE_ELEMS ((size_t)PIXELS * 32)   // one transposed plane, f16 elements
#define WS_NEEDED (3 * TPLANE_ELEMS * sizeof(_Float16))  // 12 MB

__device__ __forceinline__ int imin(int a, int b) { return a < b ? a : b; }

// ---------------- branch-free exact-ish GELU ----------------
// erf via Abramowitz-Stegun 7.1.26 (|err| <= 1.5e-7), fully branchless:
// v_rcp_f32 + 5 FMA + v_exp_f32. gelu(v) = 0.5 v (1 + erf(v/sqrt(2))).
__device__ __forceinline__ float gelu_fast(float v) {
    float z = v * 0.70710678118654752f;
    float a = __builtin_fabsf(z);
    float t = __builtin_amdgcn_rcpf(__builtin_fmaf(0.3275911f, a, 1.0f));
    float p =                 1.061405429f;
    p = __builtin_fmaf(p, t, -1.453152027f);
    p = __builtin_fmaf(p, t,  1.421413741f);
    p = __builtin_fmaf(p, t, -0.284496736f);
    p = __builtin_fmaf(p, t,  0.254829592f);
    // exp(-a^2) = exp2(-a^2 * log2(e))
    float e = __builtin_amdgcn_exp2f(a * a * -1.4426950408889634f);
    float erfa = __builtin_fmaf(-p * t, e, 1.0f);
    float erfz = __builtin_copysignf(erfa, z);
    return 0.5f * v * (1.0f + erfz);
}

// ---------------- bilinear setup ----------------
struct Bilin {
    int i00, i01, i10, i11;
    float w00, w01, w10, w11;
};

__device__ __forceinline__ Bilin bilin_setup(float gx, float gy) {
    float x = (gx + 1.0f) * 0.5f * 255.0f;
    float y = (gy + 1.0f) * 0.5f * 255.0f;
    x = fminf(fmaxf(x, 0.0f), 255.0f);
    y = fminf(fmaxf(y, 0.0f), 255.0f);
    float xf = floorf(x), yf = floorf(y);
    float wx = x - xf, wy = y - yf;
    int x0 = (int)xf, y0 = (int)yf;
    int x1 = imin(x0 + 1, RES_ - 1);
    int y1 = imin(y0 + 1, RES_ - 1);
    Bilin b;
    b.i00 = (y0 << 8) + x0; b.i01 = (y0 << 8) + x1;
    b.i10 = (y1 << 8) + x0; b.i11 = (y1 << 8) + x1;
    b.w00 = (1.0f - wx) * (1.0f - wy);
    b.w01 = wx * (1.0f - wy);
    b.w10 = (1.0f - wx) * wy;
    b.w11 = wx * wy;
    return b;
}

// Fast path: transposed f16 plane [pixel][32ch]; one 32B vector load per corner,
// packed f16 FMAs for the blend.
__device__ __forceinline__ void sample_plane_t(const _Float16* __restrict__ pT,
                                               float gx, float gy, int c0, v16h& acc) {
    Bilin b = bilin_setup(gx, gy);
    const v16h f00 = *(const v16h*)(pT + (size_t)b.i00 * 32 + c0);
    const v16h f01 = *(const v16h*)(pT + (size_t)b.i01 * 32 + c0);
    const v16h f10 = *(const v16h*)(pT + (size_t)b.i10 * 32 + c0);
    const v16h f11 = *(const v16h*)(pT + (size_t)b.i11 * 32 + c0);
    acc = acc + f00 * (_Float16)b.w00;
    acc = acc + f01 * (_Float16)b.w01;
    acc = acc + f10 * (_Float16)b.w10;
    acc = acc + f11 * (_Float16)b.w11;
}

// Slow fallback: channel-major f32 plane, scalar gathers (round-1 path).
__device__ __forceinline__ void sample_plane_f32(const float* __restrict__ plane,
                                                 float gx, float gy, int c0, float* f) {
    Bilin bl = bilin_setup(gx, gy);
    const float* pc = plane + (size_t)c0 * PLANE_CH_STRIDE;
#pragma unroll
    for (int c = 0; c < 16; ++c) {
        const float* b = pc + (size_t)c * PLANE_CH_STRIDE;
        float acc = f[c];
        acc = fmaf(b[bl.i00], bl.w00, acc);
        acc = fmaf(b[bl.i01], bl.w01, acc);
        acc = fmaf(b[bl.i10], bl.w10, acc);
        acc = fmaf(b[bl.i11], bl.w11, acc);
        f[c] = acc;
    }
}

// ---------------- WMMA operand loads ----------------
// A-operand (16x32 f16): lanes 0-15 hold K = kt*32 + {0..15},
// lanes 16-31 hold K = kt*32 + {16..31}.
__device__ __forceinline__ v16h load_A(const _Float16* actRow, int kt, int lhalf) {
    const v8h lo = *(const v8h*)(actRow + kt * 32 + lhalf * 16);
    const v8h hi = *(const v8h*)(actRow + kt * 32 + lhalf * 16 + 8);
    v16h a;
#pragma unroll
    for (int i = 0; i < 8; ++i) { a[i] = lo[i]; a[i + 8] = hi[i]; }
    return a;
}

// One 64->64 layer: D(16x64) = A(16x64) x W(64x64) + bias -> acc[4] N-tiles.
__device__ __forceinline__ void layer_gemm(const _Float16* actRow, const _Float16* wt,
                                           const float* bias, int lmod, int lhalf,
                                           v8f acc[4]) {
    v16h a0 = load_A(actRow, 0, lhalf);
    v16h a1 = load_A(actRow, 1, lhalf);
#pragma unroll
    for (int nt = 0; nt < 4; ++nt) {
        float bv = bias[nt * 16 + lmod];
        v8f c = {bv, bv, bv, bv, bv, bv, bv, bv};
        v16h b0 = *(const v16h*)(wt + (nt * 16 + lmod) * ROWS + lhalf * 16);
        v16h b1 = *(const v16h*)(wt + (nt * 16 + lmod) * ROWS + 32 + lhalf * 16);
        c = __builtin_amdgcn_wmma_f32_16x16x32_f16(false, a0, false, b0, (short)0, c, false, false);
        c = __builtin_amdgcn_wmma_f32_16x16x32_f16(false, a1, false, b1, (short)0, c, false, false);
        acc[nt] = c;
    }
}

__device__ __forceinline__ void gelu_store(const v8f acc[4], _Float16* actw,
                                           int lmod, int lhalf) {
#pragma unroll
    for (int nt = 0; nt < 4; ++nt) {
#pragma unroll
        for (int r = 0; r < 8; ++r) {
            float g = gelu_fast(acc[nt][r]);
            actw[(r + lhalf * 8) * ROWS + nt * 16 + lmod] = (_Float16)g;
        }
    }
}

// ---------------- pre-pass: transpose planes to [plane][pixel][32ch] f16 ----------------
__global__ __launch_bounds__(256) void transpose_planes_kernel(
    const float* __restrict__ pxy, const float* __restrict__ pxz,
    const float* __restrict__ pyz, _Float16* __restrict__ outT) {
    int gid = blockIdx.x * 256 + threadIdx.x;     // 3*65536 threads
    if (gid >= 3 * PIXELS) return;
    int plane = gid >> 16;
    int pix = gid & (PIXELS - 1);
    const float* src = (plane == 0) ? pxy : (plane == 1) ? pxz : pyz;
    _Float16* dst = outT + (size_t)gid * 32;
    v16h lo, hi;
#pragma unroll
    for (int c = 0; c < 16; ++c) {
        lo[c] = (_Float16)src[(size_t)c * PLANE_CH_STRIDE + pix];
        hi[c] = (_Float16)src[(size_t)(c + 16) * PLANE_CH_STRIDE + pix];
    }
    *(v16h*)dst = lo;           // 64B contiguous store
    *(v16h*)(dst + 16) = hi;
}

// ---------------- main kernel ----------------
template <bool TP>
__global__ __launch_bounds__(256) void triplane_inr_kernel(
    const float* __restrict__ coords,
    const float* __restrict__ plane_xy,
    const float* __restrict__ plane_xz,
    const float* __restrict__ plane_yz,
    const _Float16* __restrict__ planesT,
    const float* __restrict__ w0, const float* __restrict__ b0,
    const float* __restrict__ w1, const float* __restrict__ b1,
    const float* __restrict__ w2, const float* __restrict__ b2,
    const float* __restrict__ wh, const float* __restrict__ bh,
    float* __restrict__ out, int n) {

    __shared__ __align__(16) _Float16 w0t[HID * ROWS];
    __shared__ __align__(16) _Float16 w1t[HID * ROWS];
    __shared__ __align__(16) _Float16 w2t[HID * ROWS];
    __shared__ __align__(16) _Float16 act[8][16 * ROWS];  // per-wave staging
    __shared__ float b0s[HID], b1s[HID], b2s[HID], whs[HID];
    __shared__ float bhs;

    const int tid = threadIdx.x;

    // Stage weights transposed (wt[n][k]) as f16; zero-pad w0 rows k=56..63.
    for (int idx = tid; idx < HID * HID; idx += 256) {
        int nn = idx >> 6, kk = idx & 63;
        w0t[nn * ROWS + kk] = (_Float16)((kk < 56) ? w0[kk * HID + nn] : 0.0f);
        w1t[nn * ROWS + kk] = (_Float16)w1[kk * HID + nn];
        w2t[nn * ROWS + kk] = (_Float16)w2[kk * HID + nn];
    }
    if (tid < HID) {
        b0s[tid] = b0[tid];
        b1s[tid] = b1[tid];
        b2s[tid] = b2[tid];
        whs[tid] = wh[tid];
    }
    if (tid == 0) bhs = bh[0];
    __syncthreads();

    const int lane  = tid & 31;
    const int lmod  = lane & 15;   // point row within tile / N index
    const int lhalf = lane >> 4;   // channel half / lane-half
    const int waveInBlk = tid >> 5;
    _Float16* actw = &act[waveInBlk][0];

    const int numTiles = (n + 15) >> 4;
    const int waveGlob = blockIdx.x * 8 + waveInBlk;
    const int waveStep = gridDim.x * 8;
    const float bhv = bhs;

    const _Float16* pT_xy = planesT;
    const _Float16* pT_xz = planesT + TPLANE_ELEMS;
    const _Float16* pT_yz = planesT + 2 * TPLANE_ELEMS;

    for (int tile = waveGlob; tile < numTiles; tile += waveStep) {
        const int base = tile * 16;
        const int p = imin(base + lmod, n - 1);

        // ---- features: tri-plane bilinear (16 channels per lane) ----
        float cx = coords[3 * p + 0];
        float cy = coords[3 * p + 1];
        float cz = coords[3 * p + 2];
        const float inv = 2.0f / 511.0f;
        float xn = cx * inv - 1.0f;
        float yn = cy * inv - 1.0f;
        float zn = cz * inv - 1.0f;

        _Float16* row = actw + lmod * ROWS;
        const int c0 = lhalf * 16;

        if (TP) {
            v16h facc;
#pragma unroll
            for (int i = 0; i < 16; ++i) facc[i] = (_Float16)0.0f;
            sample_plane_t(pT_xy, yn, xn, c0, facc);
            sample_plane_t(pT_xz, zn, xn, c0, facc);
            sample_plane_t(pT_yz, zn, yn, c0, facc);
            // two 16B LDS stores (16B-aligned)
            *(v8h*)(row + c0)     = __builtin_shufflevector(facc, facc, 0,1,2,3,4,5,6,7);
            *(v8h*)(row + c0 + 8) = __builtin_shufflevector(facc, facc, 8,9,10,11,12,13,14,15);
        } else {
            float f[16];
#pragma unroll
            for (int c = 0; c < 16; ++c) f[c] = 0.0f;
            sample_plane_f32(plane_xy, yn, xn, c0, f);
            sample_plane_f32(plane_xz, zn, xn, c0, f);
            sample_plane_f32(plane_yz, zn, yn, c0, f);
#pragma unroll
            for (int c = 0; c < 16; ++c) row[c0 + c] = (_Float16)f[c];
        }

        if (lhalf == 0) {
            // positional encoding: k = 32 + d*8 + f (sin) / +4+f (cos)
            const float cn3[3] = {xn, yn, zn};
#pragma unroll
            for (int d = 0; d < 3; ++d) {
#pragma unroll
                for (int fq = 0; fq < 4; ++fq) {
                    float a = cn3[d] * (3.14159265358979323846f * (float)(1 << fq));
                    row[32 + d * 8 + fq]     = (_Float16)sinf(a);
                    row[32 + d * 8 + 4 + fq] = (_Float16)cosf(a);
                }
            }
        } else {
#pragma unroll
            for (int k = 56; k < 64; ++k) row[k] = (_Float16)0.0f;  // K-pad
        }
        asm volatile("s_wait_dscnt 0" ::: "memory");

        // ---- MLP via WMMA ----
        v8f acc[4];
        layer_gemm(row, w0t, b0s, lmod, lhalf, acc);
        gelu_store(acc, actw, lmod, lhalf);
        asm volatile("s_wait_dscnt 0" ::: "memory");

        layer_gemm(row, w1t, b1s, lmod, lhalf, acc);
        gelu_store(acc, actw, lmod, lhalf);
        asm volatile("s_wait_dscnt 0" ::: "memory");

        layer_gemm(row, w2t, b2s, lmod, lhalf, acc);

        // ---- head: out[m] = bh + sum_n gelu(h[m][n]) * wh[n] ----
        float partial[8];
#pragma unroll
        for (int r = 0; r < 8; ++r) partial[r] = 0.0f;
#pragma unroll
        for (int nt = 0; nt < 4; ++nt) {
            float w = whs[nt * 16 + lmod];
#pragma unroll
            for (int r = 0; r < 8; ++r)
                partial[r] = fmaf(gelu_fast(acc[nt][r]), w, partial[r]);
        }
#pragma unroll
        for (int off = 8; off >= 1; off >>= 1) {
#pragma unroll
            for (int r = 0; r < 8; ++r)
                partial[r] += __shfl_xor(partial[r], off, 16);
        }
        if (lmod == 0) {
#pragma unroll
            for (int r = 0; r < 8; ++r) {
                int idx = base + r + lhalf * 8;
                if (idx < n) out[idx] = partial[r] + bhv;
            }
        }
    }
}

extern "C" void kernel_launch(void* const* d_in, const int* in_sizes, int n_in,
                              void* d_out, int out_size, void* d_ws, size_t ws_size,
                              hipStream_t stream) {
    (void)n_in; (void)out_size;
    const float* coords   = (const float*)d_in[0];
    const float* plane_xy = (const float*)d_in[1];
    const float* plane_xz = (const float*)d_in[2];
    const float* plane_yz = (const float*)d_in[3];
    const float* w0 = (const float*)d_in[4];
    const float* b0 = (const float*)d_in[5];
    const float* w1 = (const float*)d_in[6];
    const float* b1 = (const float*)d_in[7];
    const float* w2 = (const float*)d_in[8];
    const float* b2 = (const float*)d_in[9];
    const float* wh = (const float*)d_in[10];
    const float* bh = (const float*)d_in[11];
    float* out = (float*)d_out;

    int n = in_sizes[0] / 3;
    int blocks = 2048;

    if (ws_size >= WS_NEEDED && d_ws != nullptr) {
        _Float16* planesT = (_Float16*)d_ws;
        transpose_planes_kernel<<<(3 * PIXELS + 255) / 256, 256, 0, stream>>>(
            plane_xy, plane_xz, plane_yz, planesT);
        triplane_inr_kernel<true><<<blocks, 256, 0, stream>>>(
            coords, plane_xy, plane_xz, plane_yz, planesT,
            w0, b0, w1, b1, w2, b2, wh, bh, out, n);
    } else {
        triplane_inr_kernel<false><<<blocks, 256, 0, stream>>>(
            coords, plane_xy, plane_xz, plane_yz, nullptr,
            w0, b0, w1, b1, w2, b2, wh, bh, out, n);
    }
}
